// CKAM_82291573391860
// MI455X (gfx1250) — matrix operand order repaired and checked by reading, not compiled, not run on hardware
//
#include <hip/hip_runtime.h>

// ---------------------------------------------------------------------------
// CKAM forward for MI455X (gfx1250, wave32, WMMA f32_16x16x32_f16)
//
// Shapes: B=8, C=512, CH=256, R=64, N=H*W=4096
//
// Pipeline (all linear ops algebraically fused):
//  prep:  Wc[8][64x512]  = {s_w1,s_w2,c_wq,c_wk} @ {wt,wb}[slice]   (f16)
//         bqk[4][64]     = propagated biases
//         Wf1[512x64]    = f_w[:, :256] @ s_wo (f16)
//         Wf2[512x64]    = f_w[:,256:] @ c_wo (f32, composed later with attn)
//         bf[512]        = f_w@[s_bo;c_bo] + f_b
//  k_qk:   qs,ks,qc,kc[b][64xN] = Wc_top@top + Wc_bot@bottom + bqk  (WMMA)
//          stored row-major [64,N] and transposed [N,64], f16
//  k_stats: spatial softmax row stats (max, sumexp) over m, flash pass A (WMMA)
//  k_av:    outs[r,m] = sum_n ks[r,n] * P[n,m], flash pass B (WMMA x8/iter)
//  k_ch:    channel Gram 64x64 + softmax -> attn (WMMA + LDS softmax)
//  k_w2b:   W2b[b][512x64] = Wf2 @ attn[b]  (tiny VALU)
//  k_final: out = Wf1@outsT + W2b@kcT + bf  (WMMA, K=128)
// ---------------------------------------------------------------------------

typedef __attribute__((ext_vector_type(16))) _Float16 v16h;
typedef __attribute__((ext_vector_type(8)))  _Float16 v8h;
typedef __attribute__((ext_vector_type(8)))  float    v8f;

#define NN 4096   // H*W
#define CC 512
#define CH_ 256
#define RR 64
#define BB 8

// WMMA fragment index maps (CDNA5 ISA 7.12.2, 16-bit operands, wave32):
//  A (16x32):  M = lane%16,  K(j,hi) = hi*16 + j  (contiguous order used here, not ISA table order)
//  B (32x16):  N = lane%16,  K(j,hi) = hi*16 + j  (contiguous order used here, not ISA table order)
//  D (16x16):  N = lane%16,  M(v,hi) = v + 8*hi
__device__ __forceinline__ int kmapA(int j, int hi) {
  return hi * 16 + j;
}

__device__ __forceinline__ v8f wmma16(v16h a, v16h b, v8f c) {
  return __builtin_amdgcn_wmma_f32_16x16x32_f16(false, a, false, b, (short)0, c,
                                                false, false);
}

// ---------------------------------------------------------------------------
// Weight-composition kernels (tiny; plain VALU f32)
// ---------------------------------------------------------------------------

// Wc[m][r][c] = sum_{c2<256} Wsmall[m/2][r,c2] * Wbig[m&1][(roff+c2), c]
__global__ void k_prep_qk(const float* __restrict__ wt, const float* __restrict__ wb,
                          const float* __restrict__ s_w1, const float* __restrict__ s_w2,
                          const float* __restrict__ c_wq, const float* __restrict__ c_wk,
                          _Float16* __restrict__ Wc) {
  int gid = blockIdx.x * 256 + threadIdx.x;      // 8*64*512 = 262144
  int m = gid >> 15;
  int r = (gid >> 9) & 63;
  int c = gid & 511;
  int branch = m >> 1, src = m & 1;
  const float* Ws = (branch == 0) ? s_w1 : (branch == 1) ? s_w2
                    : (branch == 2) ? c_wq : c_wk;        // [64,256]
  int roff = (branch < 2) ? 0 : 256;
  const float* Wb = src ? wb : wt;                        // [512,512]
  float acc = 0.f;
  for (int c2 = 0; c2 < 256; ++c2)
    acc += Ws[r * 256 + c2] * Wb[(size_t)(roff + c2) * 512 + c];
  Wc[gid] = (_Float16)acc;
}

__global__ void k_prep_qkb(const float* __restrict__ bt, const float* __restrict__ bb,
                           const float* __restrict__ s_w1, const float* __restrict__ s_w2,
                           const float* __restrict__ c_wq, const float* __restrict__ c_wk,
                           const float* __restrict__ s_b1, const float* __restrict__ s_b2,
                           const float* __restrict__ c_bq, const float* __restrict__ c_bk,
                           float* __restrict__ bqk) {
  int t = threadIdx.x;                 // 256 threads: branch*64 + r
  int branch = t >> 6, r = t & 63;
  const float* Ws = (branch == 0) ? s_w1 : (branch == 1) ? s_w2
                    : (branch == 2) ? c_wq : c_wk;
  const float* sb = (branch == 0) ? s_b1 : (branch == 1) ? s_b2
                    : (branch == 2) ? c_bq : c_bk;
  int roff = (branch < 2) ? 0 : 256;
  float acc = sb[r];
  for (int c2 = 0; c2 < 256; ++c2)
    acc += Ws[r * 256 + c2] * (bt[roff + c2] + bb[roff + c2]);
  bqk[t] = acc;
}

// Wf1[c,r] = f_w[c, :256] @ s_wo[:, r]  (f16)
// Wf2[c,r] = f_w[c, 256:] @ c_wo[:, r]  (f32, composed with attn later)
__global__ void k_prep_f(const float* __restrict__ f_w,
                         const float* __restrict__ s_wo, const float* __restrict__ c_wo,
                         _Float16* __restrict__ Wf1h, float* __restrict__ Wf2f) {
  int gid = blockIdx.x * 256 + threadIdx.x;      // 2*512*64 = 65536
  int sel = gid >> 15;
  int c = (gid >> 6) & 511;
  int r = gid & 63;
  const float* Wsm = sel ? c_wo : s_wo;          // [256,64]
  int coff = sel ? 256 : 0;
  float acc = 0.f;
  for (int c2 = 0; c2 < 256; ++c2)
    acc += f_w[(size_t)c * 512 + coff + c2] * Wsm[c2 * 64 + r];
  if (sel == 0) Wf1h[c * 64 + r] = (_Float16)acc;
  else          Wf2f[c * 64 + r] = acc;
}

__global__ void k_prep_fb(const float* __restrict__ f_w, const float* __restrict__ f_b,
                          const float* __restrict__ s_bo, const float* __restrict__ c_bo,
                          float* __restrict__ bfv) {
  int c = blockIdx.x * 256 + threadIdx.x;        // 512
  if (c >= 512) return;
  float acc = f_b[c];
  for (int c2 = 0; c2 < 256; ++c2)
    acc += f_w[(size_t)c * 512 + c2] * s_bo[c2]
         + f_w[(size_t)c * 512 + 256 + c2] * c_bo[c2];
  bfv[c] = acc;
}

// W2b[b][c,s] = sum_r Wf2[c,r] * attn[b][r,s]
__global__ void k_w2b(const float* __restrict__ Wf2f, const float* __restrict__ attn,
                      _Float16* __restrict__ W2b) {
  int gid = blockIdx.x * 256 + threadIdx.x;      // 8*512*64 = 262144
  int b = gid >> 15;
  int c = (gid >> 6) & 511;
  int s = gid & 63;
  float acc = 0.f;
  for (int r = 0; r < 64; ++r)
    acc += Wf2f[c * 64 + r] * attn[(size_t)b * 4096 + r * 64 + s];
  W2b[gid] = (_Float16)acc;
}

// ---------------------------------------------------------------------------
// k_qk: Out[br][b][64,N] = Wc[2br]@top[b] + Wc[2br+1]@bottom[b] + bqk[br]
// grid (32 ntiles, 4 branches, 8 batches), 256 thr = 8 waves, wave: 64r x 16n
// ---------------------------------------------------------------------------
__global__ void __launch_bounds__(256) k_qk(
    const float* __restrict__ top, const float* __restrict__ bot,
    const _Float16* __restrict__ Wc, const float* __restrict__ bqk,
    _Float16* __restrict__ out_rm, _Float16* __restrict__ out_T) {
  int tid = threadIdx.x;
  int wave = tid >> 5, lane = tid & 31, hi = lane >> 4, l16 = lane & 15;
  int br = blockIdx.y;
  int b  = blockIdx.z;
  int n0 = blockIdx.x * 128 + wave * 16;

  v8f acc[4] = {};
  for (int src = 0; src < 2; ++src) {
    const float* Xp = (src ? bot : top) + (size_t)b * CC * NN;
    const _Float16* Wm = Wc + (size_t)(br * 2 + src) * 64 * 512;
    for (int k0 = 0; k0 < 512; k0 += 32) {
      v16h bf;
#pragma unroll
      for (int j = 0; j < 16; ++j)
        bf[j] = (_Float16)Xp[(size_t)(k0 + hi * 16 + j) * NN + n0 + l16];
#pragma unroll
      for (int rt = 0; rt < 4; ++rt) {
        v16h af;
#pragma unroll
        for (int j = 0; j < 16; ++j)
          af[j] = Wm[(size_t)(rt * 16 + l16) * 512 + k0 + kmapA(j, hi)];
        acc[rt] = wmma16(af, bf, acc[rt]);
      }
    }
  }
  _Float16* orm = out_rm + ((size_t)br * BB + b) * (64 * (size_t)NN);
  _Float16* ot  = out_T  + ((size_t)br * BB + b) * ((size_t)NN * 64);
#pragma unroll
  for (int rt = 0; rt < 4; ++rt)
#pragma unroll
    for (int v = 0; v < 8; ++v) {
      int r = rt * 16 + v + 8 * hi;
      float val = acc[rt][v] + bqk[br * 64 + r];
      _Float16 h = (_Float16)val;
      orm[(size_t)r * NN + n0 + l16] = h;
      ot[(size_t)(n0 + l16) * 64 + r] = h;
    }
}

// ---------------------------------------------------------------------------
// k_stats: flash pass A. S[n,m] = sum_r qs[r,n] ks[r,m]; per-row max & sumexp.
// grid (32, 8 batches); wave owns 16 n-rows, scans all 4096 m.
// ---------------------------------------------------------------------------
__global__ void __launch_bounds__(256) k_stats(
    const _Float16* __restrict__ qsT, const _Float16* __restrict__ ksT,
    float2* __restrict__ stats) {
  int tid = threadIdx.x;
  int wave = tid >> 5, lane = tid & 31, hi = lane >> 4, l16 = lane & 15;
  int b = blockIdx.y;
  int n0 = blockIdx.x * 128 + wave * 16;
  const _Float16* qT = qsT + (size_t)b * NN * 64;
  const _Float16* kT = ksT + (size_t)b * NN * 64;

  v16h a0, a1;
#pragma unroll
  for (int j = 0; j < 16; ++j) {
    a0[j] = qT[(size_t)(n0 + l16) * 64 + kmapA(j, hi)];
    a1[j] = qT[(size_t)(n0 + l16) * 64 + 32 + kmapA(j, hi)];
  }
  float mx[8], sm[8];
#pragma unroll
  for (int v = 0; v < 8; ++v) { mx[v] = -3.0e38f; sm[v] = 0.f; }

  for (int m0 = 0; m0 < NN; m0 += 16) {
    v16h b0, b1;
#pragma unroll
    for (int j = 0; j < 16; ++j) {
      b0[j] = kT[(size_t)(m0 + l16) * 64 + hi * 16 + j];
      b1[j] = kT[(size_t)(m0 + l16) * 64 + 32 + hi * 16 + j];
    }
    v8f s = {};
    s = wmma16(a0, b0, s);
    s = wmma16(a1, b1, s);
#pragma unroll
    for (int v = 0; v < 8; ++v) {
      float x = s[v];
      float nm = fmaxf(mx[v], x);
      sm[v] = sm[v] * __expf(mx[v] - nm) + __expf(x - nm);
      mx[v] = nm;
    }
  }
  // merge across the 16 lanes holding the same row (hi bit stays fixed)
#pragma unroll
  for (int off = 1; off < 16; off <<= 1) {
#pragma unroll
    for (int v = 0; v < 8; ++v) {
      float omx = __shfl_xor(mx[v], off, 32);
      float osm = __shfl_xor(sm[v], off, 32);
      float nm = fmaxf(mx[v], omx);
      sm[v] = sm[v] * __expf(mx[v] - nm) + osm * __expf(omx - nm);
      mx[v] = nm;
    }
  }
  if (l16 == 0) {
#pragma unroll
    for (int v = 0; v < 8; ++v) {
      int n = n0 + v + 8 * hi;
      stats[(size_t)b * NN + n] = make_float2(mx[v], sm[v]);
    }
  }
}

// ---------------------------------------------------------------------------
// k_av: flash pass B. outs[r,m] = sum_n ks[r,n] * exp(S[n,m]-mx[n])/sum[n].
// grid (32, 8); wave owns m-tile of 16, accumulates all 64 r over n=0..4095.
// ---------------------------------------------------------------------------
__global__ void __launch_bounds__(256) k_av(
    const _Float16* __restrict__ qsT, const _Float16* __restrict__ ksT,
    const _Float16* __restrict__ ks_rm, const float2* __restrict__ stats,
    _Float16* __restrict__ outsT) {
  int tid = threadIdx.x;
  int wave = tid >> 5, lane = tid & 31, hi = lane >> 4, l16 = lane & 15;
  int b = blockIdx.y;
  int m0 = blockIdx.x * 128 + wave * 16;
  const _Float16* qT = qsT + (size_t)b * NN * 64;
  const _Float16* kT = ksT + (size_t)b * NN * 64;
  const _Float16* kR = ks_rm + (size_t)b * 64 * (size_t)NN;
  const float2* st = stats + (size_t)b * NN;

  // loop-invariant B operands of the score GEMM (columns m of Ks)
  v16h sb0, sb1;
#pragma unroll
  for (int j = 0; j < 16; ++j) {
    sb0[j] = kT[(size_t)(m0 + l16) * 64 + hi * 16 + j];
    sb1[j] = kT[(size_t)(m0 + l16) * 64 + 32 + hi * 16 + j];
  }
  v8f acc[4] = {};

  for (int n0 = 0; n0 < NN; n0 += 32) {
    v16h a;
    v8f s0 = {}, s1 = {};
#pragma unroll
    for (int j = 0; j < 16; ++j) a[j] = qT[(size_t)(n0 + l16) * 64 + kmapA(j, hi)];
    s0 = wmma16(a, sb0, s0);
#pragma unroll
    for (int j = 0; j < 16; ++j) a[j] = qT[(size_t)(n0 + l16) * 64 + 32 + kmapA(j, hi)];
    s0 = wmma16(a, sb1, s0);
#pragma unroll
    for (int j = 0; j < 16; ++j) a[j] = qT[(size_t)(n0 + 16 + l16) * 64 + kmapA(j, hi)];
    s1 = wmma16(a, sb0, s1);
#pragma unroll
    for (int j = 0; j < 16; ++j) a[j] = qT[(size_t)(n0 + 16 + l16) * 64 + 32 + kmapA(j, hi)];
    s1 = wmma16(a, sb1, s1);

    // normalize to probabilities (f16)
    v8h p0, p1;
#pragma unroll
    for (int v = 0; v < 8; ++v) {
      float2 t0 = st[n0 + v + 8 * hi];
      float2 t1 = st[n0 + 16 + v + 8 * hi];
      p0[v] = (_Float16)(__expf(s0[v] - t0.x) / t0.y);
      p1[v] = (_Float16)(__expf(s1[v] - t1.x) / t1.y);
    }
    // assemble B fragment (K = 32 n-values) via lane^16 exchange:
    // hi=0 lane needs partner's p0 (rows n0+8..15); hi=1 needs partner's p1.
    union U { v8h h; int i[4]; };
    U snd, rcv;
    if (hi) snd.h = p0; else snd.h = p1;
#pragma unroll
    for (int t = 0; t < 4; ++t) rcv.i[t] = __shfl_xor(snd.i[t], 16, 32);
    v8h lo, hv;
    if (hi) { lo = rcv.h; hv = p1; } else { lo = p0; hv = rcv.h; }
    v16h pb;
#pragma unroll
    for (int t = 0; t < 8; ++t) { pb[t] = lo[t]; pb[8 + t] = hv[t]; }

    // accumulate Out[r, m] += Ks[r, n-tile] @ P[n-tile, m]
#pragma unroll
    for (int rt = 0; rt < 4; ++rt) {
      v16h av;
#pragma unroll
      for (int j = 0; j < 16; ++j)
        av[j] = kR[(size_t)(rt * 16 + l16) * NN + n0 + kmapA(j, hi)];
      acc[rt] = wmma16(av, pb, acc[rt]);
    }
  }
  _Float16* oT = outsT + (size_t)b * NN * 64;
#pragma unroll
  for (int rt = 0; rt < 4; ++rt)
#pragma unroll
    for (int v = 0; v < 8; ++v) {
      int r = rt * 16 + v + 8 * hi;
      oT[(size_t)(m0 + l16) * 64 + r] = (_Float16)acc[rt][v];
    }
}

// ---------------------------------------------------------------------------
// k_ch: channel Gram[64,64] = qc @ kc^T over N, softmax rows -> attn (f32)
// grid (8 batches), 256 thr; wave w: rtile = w&3, stiles = {(w>>2)*2, +1}
// ---------------------------------------------------------------------------
__global__ void __launch_bounds__(256) k_ch(
    const _Float16* __restrict__ qc_rm, const _Float16* __restrict__ kc_rm,
    float* __restrict__ attn) {
  __shared__ float smem[64 * 64];
  int tid = threadIdx.x;
  int wave = tid >> 5, lane = tid & 31, hi = lane >> 4, l16 = lane & 15;
  int b = blockIdx.x;
  int rt = wave & 3, stb = (wave >> 2) * 2;
  const _Float16* q = qc_rm + (size_t)b * 64 * (size_t)NN;
  const _Float16* k = kc_rm + (size_t)b * 64 * (size_t)NN;

  v8f g0 = {}, g1 = {};
  for (int n0 = 0; n0 < NN; n0 += 32) {
    v16h a;
#pragma unroll
    for (int j = 0; j < 16; ++j)
      a[j] = q[(size_t)(rt * 16 + l16) * NN + n0 + kmapA(j, hi)];
    v16h b0, b1;
#pragma unroll
    for (int j = 0; j < 16; ++j) {
      b0[j] = k[(size_t)((stb + 0) * 16 + l16) * NN + n0 + hi * 16 + j];
      b1[j] = k[(size_t)((stb + 1) * 16 + l16) * NN + n0 + hi * 16 + j];
    }
    g0 = wmma16(a, b0, g0);
    g1 = wmma16(a, b1, g1);
  }
#pragma unroll
  for (int v = 0; v < 8; ++v) {
    int r = rt * 16 + v + 8 * hi;
    smem[r * 64 + (stb + 0) * 16 + l16] = g0[v];
    smem[r * 64 + (stb + 1) * 16 + l16] = g1[v];
  }
  __syncthreads();
  if (tid < 64) {
    int r = tid;
    float m = -3.0e38f;
    for (int s = 0; s < 64; ++s) m = fmaxf(m, smem[r * 64 + s]);
    float su = 0.f;
    for (int s = 0; s < 64; ++s) su += __expf(smem[r * 64 + s] - m);
    float inv = 1.0f / su;
    for (int s = 0; s < 64; ++s)
      attn[(size_t)b * 4096 + r * 64 + s] = __expf(smem[r * 64 + s] - m) * inv;
  }
}

// ---------------------------------------------------------------------------
// k_final: out[b][c,n] = Wf1@outsT + W2b[b]@kcT + bf   (K = 64+64)
// grid (64 ncol, 16 mrow, 8 batch); wave -> one 16x16 tile.
// ---------------------------------------------------------------------------
__global__ void __launch_bounds__(256) k_final(
    const _Float16* __restrict__ Wf1h, const _Float16* __restrict__ W2b,
    const _Float16* __restrict__ outsT, const _Float16* __restrict__ kcT,
    const float* __restrict__ bfv, float* __restrict__ out) {
  int tid = threadIdx.x;
  int wave = tid >> 5, lane = tid & 31, hi = lane >> 4, l16 = lane & 15;
  int b = blockIdx.z;
  int m0 = blockIdx.y * 32 + (wave >> 2) * 16;
  int n0 = blockIdx.x * 64 + (wave & 3) * 16;
  const _Float16* oT = outsT + (size_t)b * NN * 64;
  const _Float16* kT = kcT + (size_t)b * NN * 64;
  const _Float16* W2 = W2b + (size_t)b * 512 * 64;

  v8f acc = {};
#pragma unroll
  for (int k0 = 0; k0 < 64; k0 += 32) {
    v16h a, bf;
#pragma unroll
    for (int j = 0; j < 16; ++j) {
      a[j]  = Wf1h[(size_t)(m0 + l16) * 64 + k0 + kmapA(j, hi)];
      bf[j] = oT[(size_t)(n0 + l16) * 64 + k0 + hi * 16 + j];
    }
    acc = wmma16(a, bf, acc);
#pragma unroll
    for (int j = 0; j < 16; ++j) {
      a[j]  = W2[(size_t)(m0 + l16) * 64 + k0 + kmapA(j, hi)];
      bf[j] = kT[(size_t)(n0 + l16) * 64 + k0 + hi * 16 + j];
    }
    acc = wmma16(a, bf, acc);
  }
#pragma unroll
  for (int v = 0; v < 8; ++v) {
    int r = m0 + v + 8 * hi;
    out[((size_t)b * 512 + r) * NN + n0 + l16] = acc[v] + bfv[r];
  }
}

// ---------------------------------------------------------------------------
extern "C" void kernel_launch(void* const* d_in, const int* in_sizes, int n_in,
                              void* d_out, int out_size, void* d_ws, size_t ws_size,
                              hipStream_t stream) {
  (void)in_sizes; (void)n_in; (void)out_size; (void)ws_size;
  const float* top  = (const float*)d_in[0];
  const float* bot  = (const float*)d_in[1];
  const float* wt   = (const float*)d_in[2];
  const float* bt   = (const float*)d_in[3];
  const float* wb   = (const float*)d_in[4];
  const float* bb   = (const float*)d_in[5];
  const float* s_w1 = (const float*)d_in[6];
  const float* s_b1 = (const float*)d_in[7];
  const float* s_w2 = (const float*)d_in[8];
  const float* s_b2 = (const float*)d_in[9];
  const float* s_wo = (const float*)d_in[10];
  const float* s_bo = (const float*)d_in[11];
  const float* c_wq = (const float*)d_in[12];
  const float* c_bq = (const float*)d_in[13];
  const float* c_wk = (const float*)d_in[14];
  const float* c_bk = (const float*)d_in[15];
  const float* c_wo = (const float*)d_in[16];
  const float* c_bo = (const float*)d_in[17];
  const float* f_w  = (const float*)d_in[18];
  const float* f_b  = (const float*)d_in[19];
  float* out = (float*)d_out;

  char* ws = (char*)d_ws;
  size_t off = 0;
  auto take = [&](size_t bytes) {
    size_t o = off;
    off += (bytes + 255) & ~(size_t)255;
    return o;
  };
  _Float16* Wc    = (_Float16*)(ws + take(8ull * 64 * 512 * 2));
  _Float16* Wf1h  = (_Float16*)(ws + take(512ull * 64 * 2));
  float*    Wf2f  = (float*)   (ws + take(512ull * 64 * 4));
  float*    bqk   = (float*)   (ws + take(4ull * 64 * 4));
  float*    bfv   = (float*)   (ws + take(512ull * 4));
  _Float16* rm    = (_Float16*)(ws + take(4ull * BB * 64 * NN * 2));  // [br][b][64][N]
  _Float16* tT    = (_Float16*)(ws + take(4ull * BB * NN * 64 * 2));  // [br][b][N][64]
  float2*   stats = (float2*)  (ws + take((size_t)BB * NN * 8));
  _Float16* outsT = (_Float16*)(ws + take((size_t)BB * NN * 64 * 2));
  float*    attn  = (float*)   (ws + take((size_t)BB * 64 * 64 * 4));
  _Float16* W2b   = (_Float16*)(ws + take((size_t)BB * 512 * 64 * 2));

  const size_t MATQ = (size_t)BB * 64 * NN;  // elements per branch (both layouts)

  k_prep_qk <<<dim3(1024), dim3(256), 0, stream>>>(wt, wb, s_w1, s_w2, c_wq, c_wk, Wc);
  k_prep_qkb<<<dim3(1),    dim3(256), 0, stream>>>(bt, bb, s_w1, s_w2, c_wq, c_wk,
                                                   s_b1, s_b2, c_bq, c_bk, bqk);
  k_prep_f  <<<dim3(256),  dim3(256), 0, stream>>>(f_w, s_wo, c_wo, Wf1h, Wf2f);
  k_prep_fb <<<dim3(2),    dim3(256), 0, stream>>>(f_w, f_b, s_bo, c_bo, bfv);

  // branches: 0=qs 1=ks 2=qc 3=kc
  k_qk   <<<dim3(32, 4, 8), dim3(256), 0, stream>>>(top, bot, Wc, bqk, rm, tT);
  k_stats<<<dim3(32, 8),    dim3(256), 0, stream>>>(tT /*qsT*/, tT + MATQ /*ksT*/, stats);
  k_av   <<<dim3(32, 8),    dim3(256), 0, stream>>>(tT, tT + MATQ, rm + MATQ /*ks rm*/,
                                                    stats, outsT);
  k_ch   <<<dim3(8),        dim3(256), 0, stream>>>(rm + 2 * MATQ /*qc*/,
                                                    rm + 3 * MATQ /*kc*/, attn);
  k_w2b  <<<dim3(1024),     dim3(256), 0, stream>>>(Wf2f, attn, W2b);
  k_final<<<dim3(64, 16, 8), dim3(256), 0, stream>>>(Wf1h, W2b, outsT,
                                                     tT + 3 * MATQ /*kcT*/, bfv, out);
}
